// LocMotionAppearance_9053791060039
// MI455X (gfx1250) — matrix operand order repaired and checked by reading, not compiled, not run on hardware
//
#include <hip/hip_runtime.h>
#include <hip/hip_bf16.h>
#include <math.h>

// ---------------------------------------------------------------------------
// Problem constants (match reference: B=8, HW=512, S=2000)
// ---------------------------------------------------------------------------
#define B_    8
#define HW_   512
#define S_    2000
#define NSEG  (B_ * S_)          // 16000 tokens
#define NPIX  (B_ * HW_ * HW_)   // 2,097,152 pixels
#define EPSV  1e-12f

#define KC     128               // K chunk staged in LDS (double buffered)
#define BPITCH (KC + 8)          // LDS row pitch in f16 (272 B: 16B-aligned,
                                 // rotates banks by 4/row -> conflict-free b128)
// TDM pad codes for the 272B pitch: pad 4 DWORDs (code 3) every 64 DWORDs (code 5)
#define PAD_INTERVAL_CODE 5
#define PAD_AMOUNT_CODE   3

#define DEVINL __device__ __forceinline__

typedef __attribute__((ext_vector_type(16))) _Float16 v16h;
typedef __attribute__((ext_vector_type(8)))  _Float16 v8h;
typedef __attribute__((ext_vector_type(8)))  float    v8f;

union V16U { v16h v; v8h h[2]; };

#if __has_builtin(__builtin_amdgcn_tensor_load_to_lds)
#define USE_TDM 1
#else
#define USE_TDM 0
#endif

DEVINL v16h frag_zero() {
    v16h z;
#pragma unroll
    for (int i = 0; i < 16; ++i) z[i] = (_Float16)0.0f;
    return z;
}

DEVINL float celu1(float x) { return x > 0.0f ? x : (__expf(x) - 1.0f); }

DEVINL float wave_sum(float v) {
#pragma unroll
    for (int off = 16; off > 0; off >>= 1) v += __shfl_xor(v, off);
    return v;
}

// A fragment (16x32 f16): lane l holds row (l&15), K = kb..kb+7 and
// kb+16..kb+23 with kb = (l>=16 ? 8 : 0). SHIFT=true adds the conv1d row
// shift with zero padding; SHIFT=false is branch-free.
template<bool SHIFT>
DEVINL v16h load_a(const _Float16* __restrict__ A, int ld, int row0, int k0,
                   int shift, int M) {
    const int lane = threadIdx.x & 31;
    int r = row0 + (lane & 15);
    if (SHIFT) {
        r += shift;
        if (r < 0 || r >= M) return frag_zero();
    }
    const int kb = k0 + ((lane >> 4) << 3);
    const _Float16* p = A + (size_t)r * ld + kb;
    V16U u;
    u.h[0] = *(const v8h*)(p);
    u.h[1] = *(const v8h*)(p + 16);
    return u.v;
}

// B fragment (32x16 f16) from the LDS-staged weight tile [64][BPITCH]:
// lane l holds output col (l&15) of 16-col group nf, K = kb..kb+7 and
// kb+16..kb+23 with kb = k0 + (l>=16 ? 8 : 0). Two ds_load_b128.
DEVINL v16h load_b_lds(const _Float16* sB, int nf, int k0) {
    const int lane = threadIdx.x & 31;
    const _Float16* p = sB + (size_t)(nf * 16 + (lane & 15)) * BPITCH
                           + k0 + ((lane >> 4) << 3);
    V16U u;
    u.h[0] = *(const v8h*)(p);
    u.h[1] = *(const v8h*)(p + 16);
    return u.v;
}

// ---------------------------------------------------------------------------
// TDM staging of a 64-row x KC weight tile into LDS (descriptor per
// CDNA5 ISA; 2D tile, data_size=2B, padded LDS rows).
// ---------------------------------------------------------------------------
#if USE_TDM
typedef __attribute__((ext_vector_type(4))) unsigned int u32x4;
typedef __attribute__((ext_vector_type(8))) int          i32x8;
typedef __attribute__((ext_vector_type(4))) int          i32x4;

DEVINL void tdm_load_b_tile(unsigned lds_off, const _Float16* gsrc, int ldK) {
    unsigned long long ga = (unsigned long long)(size_t)gsrc;    // byte address
    u32x4 g0;
    g0.x = 1u;                                     // count=1 (user descriptor)
    g0.y = lds_off;                                // lds_addr (bytes)
    g0.z = (unsigned)(ga & 0xFFFFFFFFull);         // global_addr[31:0]
    g0.w = (unsigned)((ga >> 32) & 0x01FFFFFFull)  // global_addr[56:32]
         | 0x80000000u;                            // type=2 ("image")
    i32x8 g1;
    g1[0] = (1 << 16)                     // data_size = 2 bytes
          | (1 << 20)                     // pad_enable
          | (PAD_INTERVAL_CODE << 22)     // pad every 64 DWORDs (256 B)
          | (PAD_AMOUNT_CODE   << 25);    // pad 4 DWORDs (16 B) -> 272 B pitch
    g1[1] = (KC & 0xFFFF) << 16;                   // tensor_dim0[15:0]
    g1[2] = (KC >> 16) | ((64 & 0xFFFF) << 16);    // tensor_dim0 hi | tensor_dim1 lo
    g1[3] = (64 >> 16) | ((KC & 0xFFFF) << 16);    // tensor_dim1 hi | tile_dim0
    g1[4] = 64;                                    // tile_dim1=64, tile_dim2=0
    g1[5] = ldK;                                   // tensor_dim0_stride[31:0] (elems)
    g1[6] = 0;                                     // stride hi | dim1_stride lo
    g1[7] = 0;
    i32x4 z4 = {0, 0, 0, 0};
#if __clang_major__ >= 23
    i32x8 z8 = {0, 0, 0, 0, 0, 0, 0, 0};
    __builtin_amdgcn_tensor_load_to_lds(g0, g1, z4, z4, z8, 0);
#else
    __builtin_amdgcn_tensor_load_to_lds(g0, g1, z4, z4, 0);
#endif
}

DEVINL void tdm_wait() {
#if __has_builtin(__builtin_amdgcn_s_wait_tensorcnt)
    __builtin_amdgcn_s_wait_tensorcnt(0);
#else
    asm volatile("s_wait_tensorcnt 0x0" ::: "memory");
#endif
}
#endif

// ---------------------------------------------------------------------------
// 0. zero fill
// ---------------------------------------------------------------------------
__global__ void k_zero(float* __restrict__ p, int n) {
    int i = blockIdx.x * blockDim.x + threadIdx.x;
    if (i < n) p[i] = 0.0f;
}

// ---------------------------------------------------------------------------
// 1. superpixel pooling (segment sums via global fp32 atomics)
// ---------------------------------------------------------------------------
__global__ void k_pool(const int* __restrict__ labels, const float* __restrict__ fx,
                       const float* __restrict__ fy, float* __restrict__ sums,
                       float* __restrict__ cnt) {
    int i = blockIdx.x * blockDim.x + threadIdx.x;
    if (i >= NPIX) return;
    int b   = i / (HW_ * HW_);
    int rem = i - b * (HW_ * HW_);
    int w   = rem / HW_;
    int h   = rem - w * HW_;
    int seg = b * S_ + labels[i];
    float xx = (float)w * (1.0f / (float)(HW_ - 1));
    float yy = (float)h * (1.0f / (float)(HW_ - 1));
    atomicAdd(&sums[seg * 4 + 0], xx);
    atomicAdd(&sums[seg * 4 + 1], yy);
    atomicAdd(&sums[seg * 4 + 2], fx[i]);
    atomicAdd(&sums[seg * 4 + 3], fy[i]);
    atomicAdd(&cnt[seg], 1.0f);
}

// ---------------------------------------------------------------------------
// 2. build x5 = [t, mean_x, mean_y, mean_fx, mean_fy]  (fp32, [NSEG,5])
// ---------------------------------------------------------------------------
__global__ void k_build_x5(const float* __restrict__ sums, const float* __restrict__ cnt,
                           const int* __restrict__ fidx, const int* __restrict__ nfr,
                           float* __restrict__ x5) {
    int n = blockIdx.x * blockDim.x + threadIdx.x;
    if (n >= NSEG) return;
    int b = n / S_;
    float t   = (float)fidx[b] / ((float)nfr[b] - 1.0f);
    float inv = 1.0f / fmaxf(cnt[n], 1.0f);
    x5[n * 5 + 0] = t;
    x5[n * 5 + 1] = sums[n * 4 + 0] * inv;
    x5[n * 5 + 2] = sums[n * 4 + 1] * inv;
    x5[n * 5 + 3] = sums[n * 4 + 2] * inv;
    x5[n * 5 + 4] = sums[n * 4 + 3] * inv;
}

// ---------------------------------------------------------------------------
// 3. weight preprocessing
// ---------------------------------------------------------------------------
__global__ void k_f32_to_f16(const float* __restrict__ src, _Float16* __restrict__ dst, int n) {
    int i = blockIdx.x * blockDim.x + threadIdx.x;
    if (i < n) dst[i] = (_Float16)src[i];
}

// convm_w [512,512,3] -> three contiguous f16 [512,512] (one per tap)
__global__ void k_conv_split(const float* __restrict__ w, _Float16* __restrict__ wc) {
    int i = blockIdx.x * blockDim.x + threadIdx.x;     // i indexes (cout, cin, s)
    if (i >= 512 * 512 * 3) return;
    int s    = i % 3;
    int cin  = (i / 3) % 512;
    int cout = i / (3 * 512);
    wc[(size_t)s * 512 * 512 + (size_t)cout * 512 + cin] = (_Float16)w[i];
}

// dist_w [15,128] -> row-l2-normalized fp32
__global__ void k_dist_norm(const float* __restrict__ dw, float* __restrict__ wn) {
    __shared__ float red[128];
    int r = blockIdx.x, t = threadIdx.x;
    float v = dw[r * 128 + t];
    red[t] = v * v;
    __syncthreads();
    for (int s = 64; s > 0; s >>= 1) {
        if (t < s) red[t] += red[t + s];
        __syncthreads();
    }
    wn[r * 128 + t] = v / fmaxf(sqrtf(red[0]), EPSV);
}

// ---------------------------------------------------------------------------
// 4. conv1 (Cin=5, K=3, pad=1) + celu -> f16 [NSEG, 256]
// ---------------------------------------------------------------------------
__global__ void k_conv1(const float* __restrict__ x5, const float* __restrict__ w,
                        const float* __restrict__ b, _Float16* __restrict__ xa) {
    __shared__ float xs[3][5];
    int n = blockIdx.x, t = threadIdx.x;
    if (t < 15) {
        int k = t / 5, ci = t % 5;
        int m = n + k - 1;
        xs[k][ci] = (m >= 0 && m < NSEG) ? x5[m * 5 + ci] : 0.0f;
    }
    __syncthreads();
    float acc = b[t];
#pragma unroll
    for (int ci = 0; ci < 5; ++ci)
#pragma unroll
        for (int k = 0; k < 3; ++k)
            acc += xs[k][ci] * w[(t * 5 + ci) * 3 + k];
    xa[(size_t)n * 256 + t] = (_Float16)celu1(acc);
}

// ---------------------------------------------------------------------------
// 5. WMMA GEMM, TDM double-buffered LDS weight staging.
//    C[M,N] = celu?( bias + sum_tap A[n+shift(tap), :] @ Wtap[N,K]^T )
//    128 threads = 4 waves; block tile 128 rows x 64 cols; wave 32x64
//    (2 A-frags x 4 B-frags = 8 WMMAs per 32-wide k-step).
//    Pipeline: wait chunk c -> barrier -> issue TDM for chunk c+1 into the
//    other buffer -> compute chunk c (DMA overlaps compute).
// ---------------------------------------------------------------------------
template<int TAPS>
__global__ void k_gemm(const _Float16* __restrict__ A, const _Float16* __restrict__ W,
                       const float* __restrict__ bias, _Float16* __restrict__ outH,
                       float* __restrict__ outF, int M, int N, int K,
                       long long wTapStride, int doCelu) {
    __shared__ _Float16 sB[2][64 * BPITCH];       // ~34 KB, double buffered

    const int lane = threadIdx.x & 31;
    const int wid  = threadIdx.x >> 5;
    const int row0 = blockIdx.x * 128 + wid * 32;
    const int n0   = blockIdx.y * 64;

    const int chunksPerTap = K / KC;
    const int totalChunks  = TAPS * chunksPerTap;

    v8f acc0[4], acc1[4];
#pragma unroll
    for (int nf = 0; nf < 4; ++nf)
#pragma unroll
        for (int j = 0; j < 8; ++j) { acc0[nf][j] = 0.0f; acc1[nf][j] = 0.0f; }

    auto stageChunk = [&](int c, int sel) {
        const int tap = c / chunksPerTap;
        const int kc  = (c - tap * chunksPerTap) * KC;
        const _Float16* Wt = W + (size_t)tap * wTapStride;
#if USE_TDM
        if (threadIdx.x < 32)   // one wave drives the DMA
            tdm_load_b_tile((unsigned)(size_t)&sB[sel][0],
                            Wt + (size_t)n0 * K + kc, K);
#else
        for (int i = threadIdx.x; i < 64 * (KC / 8); i += 128) {
            int r  = i / (KC / 8);
            int c8 = (i % (KC / 8)) * 8;
            *(v8h*)(&sB[sel][0] + (size_t)r * BPITCH + c8) =
                *(const v8h*)(Wt + (size_t)(n0 + r) * K + kc + c8);
        }
#endif
    };

    stageChunk(0, 0);

    for (int c = 0; c < totalChunks; ++c) {
        const int sel   = c & 1;
        const int tap   = c / chunksPerTap;
        const int kc    = (c - tap * chunksPerTap) * KC;
        const int shift = tap - (TAPS >> 1);

#if USE_TDM
        if (threadIdx.x < 32) tdm_wait();          // chunk c landed
#endif
        __syncthreads();                           // visible to all; buf sel^1 free
        if (c + 1 < totalChunks) stageChunk(c + 1, sel ^ 1);   // DMA overlaps compute

        // prefetch the next A chunk for this block's 128 rows (global_prefetch_b8)
        {
            int cn = (c + 1 < totalChunks) ? c + 1 : c;
            int tn = cn / chunksPerTap;
            int kn = (cn - tn * chunksPerTap) * KC;
            int pr = blockIdx.x * 128 + (int)threadIdx.x;
            __builtin_prefetch(A + (size_t)pr * K + kn, 0, 1);
        }

        const _Float16* sBc = &sB[sel][0];
#pragma unroll
        for (int kk = 0; kk < KC; kk += 32) {
            v16h a0 = load_a<(TAPS != 1)>(A, K, row0,      kc + kk, shift, M);
            v16h a1 = load_a<(TAPS != 1)>(A, K, row0 + 16, kc + kk, shift, M);
            v16h b0 = load_b_lds(sBc, 0, kk);
            v16h b1 = load_b_lds(sBc, 1, kk);
            v16h b2 = load_b_lds(sBc, 2, kk);
            v16h b3 = load_b_lds(sBc, 3, kk);
            acc0[0] = __builtin_amdgcn_wmma_f32_16x16x32_f16(false, a0, false, b0, (short)0, acc0[0], false, false);
            acc1[0] = __builtin_amdgcn_wmma_f32_16x16x32_f16(false, a1, false, b0, (short)0, acc1[0], false, false);
            acc0[1] = __builtin_amdgcn_wmma_f32_16x16x32_f16(false, a0, false, b1, (short)0, acc0[1], false, false);
            acc1[1] = __builtin_amdgcn_wmma_f32_16x16x32_f16(false, a1, false, b1, (short)0, acc1[1], false, false);
            acc0[2] = __builtin_amdgcn_wmma_f32_16x16x32_f16(false, a0, false, b2, (short)0, acc0[2], false, false);
            acc1[2] = __builtin_amdgcn_wmma_f32_16x16x32_f16(false, a1, false, b2, (short)0, acc1[2], false, false);
            acc0[3] = __builtin_amdgcn_wmma_f32_16x16x32_f16(false, a0, false, b3, (short)0, acc0[3], false, false);
            acc1[3] = __builtin_amdgcn_wmma_f32_16x16x32_f16(false, a1, false, b3, (short)0, acc1[3], false, false);
        }
    }

    const int col_lane = lane & 15;
    const int rshift   = (lane >> 4) << 3;
#pragma unroll
    for (int nf = 0; nf < 4; ++nf) {
        int col  = n0 + nf * 16 + col_lane;
        float bv = bias[col];
#pragma unroll
        for (int j = 0; j < 8; ++j) {
            float v0 = acc0[nf][j] + bv;
            float v1 = acc1[nf][j] + bv;
            if (doCelu) { v0 = celu1(v0); v1 = celu1(v1); }
            size_t i0 = (size_t)(row0 + rshift + j) * N + col;
            size_t i1 = (size_t)(row0 + 16 + rshift + j) * N + col;
            if (outF) { outF[i0] = v0;            outF[i1] = v1; }
            else      { outH[i0] = (_Float16)v0;  outH[i1] = (_Float16)v1; }
        }
    }
}

// ---------------------------------------------------------------------------
// 6. l2norm over 256 lin1 features + concat dec_feats -> Xcat f16 [NSEG,512]
// ---------------------------------------------------------------------------
__global__ void k_l2cat(const _Float16* __restrict__ L, const float* __restrict__ dec,
                        _Float16* __restrict__ X) {
    __shared__ float red[256];
    int row = blockIdx.x, t = threadIdx.x;
    float v = (float)L[(size_t)row * 256 + t];
    red[t] = v * v;
    __syncthreads();
    for (int s = 128; s > 0; s >>= 1) {
        if (t < s) red[t] += red[t + s];
        __syncthreads();
    }
    float inv = 1.0f / fmaxf(sqrtf(red[0]), EPSV);
    X[(size_t)row * 512 + t]       = (_Float16)(v * inv);
    X[(size_t)row * 512 + 256 + t] = (_Float16)dec[(size_t)row * 256 + t];
}

// ---------------------------------------------------------------------------
// 7. heads: rho = h@pred_w^T + b (pre-norm), xn = l2norm(h), cs = xn@wn^T
// ---------------------------------------------------------------------------
__global__ void k_head(const float* __restrict__ H, const float* __restrict__ pw,
                       const float* __restrict__ pb, const float* __restrict__ wn,
                       float* __restrict__ out_cs, float* __restrict__ out_x,
                       float* __restrict__ out_rho) {
    int lane = threadIdx.x & 31;
    int row  = blockIdx.x * 8 + (threadIdx.x >> 5);
    const float* h = H + (size_t)row * 128;

    float hl[4];
#pragma unroll
    for (int i = 0; i < 4; ++i) hl[i] = h[lane * 4 + i];

    float ss = 0.0f, pr = 0.0f;
#pragma unroll
    for (int i = 0; i < 4; ++i) {
        ss += hl[i] * hl[i];
        pr += hl[i] * pw[lane * 4 + i];
    }
    ss = wave_sum(ss);
    pr = wave_sum(pr);
    float inv = 1.0f / fmaxf(sqrtf(ss), EPSV);
    if (lane == 0) out_rho[row] = pr + pb[0];

    float xn[4];
#pragma unroll
    for (int i = 0; i < 4; ++i) {
        xn[i] = hl[i] * inv;
        out_x[(size_t)row * 128 + lane * 4 + i] = xn[i];
    }
#pragma unroll
    for (int j = 0; j < 15; ++j) {
        float p = 0.0f;
#pragma unroll
        for (int i = 0; i < 4; ++i) p += xn[i] * wn[j * 128 + lane * 4 + i];
        p = wave_sum(p);
        if (lane == 0) out_cs[(size_t)row * 15 + j] = p;
    }
}

// ---------------------------------------------------------------------------
// launch
// ---------------------------------------------------------------------------
extern "C" void kernel_launch(void* const* d_in, const int* in_sizes, int n_in,
                              void* d_out, int out_size, void* d_ws, size_t ws_size,
                              hipStream_t stream) {
    const int*   labels   = (const int*)  d_in[0];
    const float* fx       = (const float*)d_in[1];
    const float* fy       = (const float*)d_in[2];
    const int*   fidx     = (const int*)  d_in[3];
    const int*   nfr      = (const int*)  d_in[4];
    const float* dec      = (const float*)d_in[5];
    const float* conv1_w  = (const float*)d_in[6];
    const float* conv1_b  = (const float*)d_in[7];
    const float* lin1_w   = (const float*)d_in[8];
    const float* lin1_b   = (const float*)d_in[9];
    const float* convm_w  = (const float*)d_in[10];
    const float* convm_b  = (const float*)d_in[11];
    const float* linm1_w  = (const float*)d_in[12];
    const float* linm1_b  = (const float*)d_in[13];
    const float* linm2_w  = (const float*)d_in[14];
    const float* linm2_b  = (const float*)d_in[15];
    const float* pred_w   = (const float*)d_in[16];
    const float* pred_b   = (const float*)d_in[17];
    const float* dist_w   = (const float*)d_in[18];

    char* ws = (char*)d_ws;
    size_t off = 0;
    auto alloc = [&](size_t bytes) -> char* {
        char* p = ws + off;
        off += (bytes + 255) & ~(size_t)255;
        return p;
    };

    float*    sums = (float*)   alloc((size_t)NSEG * 4 * 4);
    float*    cnt  = (float*)   alloc((size_t)NSEG * 4);
    float*    x5   = (float*)   alloc((size_t)NSEG * 5 * 4);
    _Float16* wl1  = (_Float16*)alloc((size_t)256 * 256 * 2);
    _Float16* wch  = (_Float16*)alloc((size_t)3 * 512 * 512 * 2);
    _Float16* wm1  = (_Float16*)alloc((size_t)256 * 512 * 2);
    _Float16* wm2  = (_Float16*)alloc((size_t)128 * 256 * 2);
    float*    wn   = (float*)   alloc((size_t)15 * 128 * 4);
    _Float16* xa   = (_Float16*)alloc((size_t)NSEG * 256 * 2);  // conv1 out
    _Float16* L    = (_Float16*)alloc((size_t)NSEG * 256 * 2);  // lin1 out, reused as linm1 out
    _Float16* Xc   = (_Float16*)alloc((size_t)NSEG * 512 * 2);  // l2norm ++ dec
    _Float16* Y    = (_Float16*)alloc((size_t)NSEG * 512 * 2);  // convm out
    float*    H2   = (float*)   alloc((size_t)NSEG * 128 * 4);  // linm2 out (fp32)

    float* out_cs  = (float*)d_out;
    float* out_x   = out_cs + (size_t)NSEG * 15;
    float* out_rho = out_x  + (size_t)NSEG * 128;

    // 1. pooling (zero accumulators every call — deterministic under replay)
    k_zero<<<(NSEG * 4 + 255) / 256, 256, 0, stream>>>(sums, NSEG * 4);
    k_zero<<<(NSEG + 255) / 256, 256, 0, stream>>>(cnt, NSEG);
    k_pool<<<NPIX / 256, 256, 0, stream>>>(labels, fx, fy, sums, cnt);
    k_build_x5<<<(NSEG + 255) / 256, 256, 0, stream>>>(sums, cnt, fidx, nfr, x5);

    // 2. weight prep
    k_f32_to_f16<<<(256 * 256 + 255) / 256, 256, 0, stream>>>(lin1_w, wl1, 256 * 256);
    k_f32_to_f16<<<(256 * 512 + 255) / 256, 256, 0, stream>>>(linm1_w, wm1, 256 * 512);
    k_f32_to_f16<<<(128 * 256 + 255) / 256, 256, 0, stream>>>(linm2_w, wm2, 128 * 256);
    k_conv_split<<<(512 * 512 * 3 + 255) / 256, 256, 0, stream>>>(convm_w, wch);
    k_dist_norm<<<15, 128, 0, stream>>>(dist_w, wn);

    // 3. conv1 + celu
    k_conv1<<<NSEG, 256, 0, stream>>>(x5, conv1_w, conv1_b, xa);

    // 4. lin1 + celu  (M=16000, N=256, K=256)
    k_gemm<1><<<dim3(NSEG / 128, 256 / 64), 128, 0, stream>>>(
        xa, wl1, lin1_b, L, (float*)nullptr, NSEG, 256, 256, 0, 1);

    // 5. l2norm + concat dec_feats
    k_l2cat<<<NSEG, 256, 0, stream>>>(L, dec, Xc);

    // 6. convm (3-tap) + celu  (M=16000, N=512, K=3x512)
    k_gemm<3><<<dim3(NSEG / 128, 512 / 64), 128, 0, stream>>>(
        Xc, wch, convm_b, Y, (float*)nullptr, NSEG, 512, 512, 512LL * 512, 1);

    // 7. linm1 + celu  (M=16000, N=256, K=512)
    k_gemm<1><<<dim3(NSEG / 128, 256 / 64), 128, 0, stream>>>(
        Y, wm1, linm1_b, L, (float*)nullptr, NSEG, 256, 512, 0, 1);

    // 8. linm2 + celu -> fp32  (M=16000, N=128, K=256)
    k_gemm<1><<<dim3(NSEG / 128, 128 / 64), 128, 0, stream>>>(
        L, wm2, linm2_b, (_Float16*)nullptr, H2, NSEG, 128, 256, 0, 1);

    // 9. heads
    k_head<<<NSEG / 8, 256, 0, stream>>>(H2, pred_w, pred_b, wn, out_cs, out_x, out_rho);

    (void)in_sizes; (void)n_in; (void)out_size; (void)ws_size;
}
